// GATTransfomerOnlyDecoder_44641890075099
// MI455X (gfx1250) — matrix operand order repaired and checked by reading, not compiled, not run on hardware
//
#include <hip/hip_runtime.h>
#include <hip/hip_bf16.h>
#include <math.h>

// ---------------- problem constants ----------------------------------------
static constexpr int kN   = 4096;
static constexpr int kE   = 131072;
static constexpr int kNF  = 16;
static constexpr int kEMB = 64;
static constexpr int kH   = 4;
static constexpr int kC1  = 64;
static constexpr int kC2  = 32;
static constexpr int kFF  = 512;
static constexpr int kOUT = 8;
static constexpr int kNG  = 8;
static constexpr int kD   = 128;
static constexpr int kDH  = 32;
static constexpr int kMSG = kE + kN;

typedef __bf16 bhalf;
typedef __attribute__((ext_vector_type(16))) __bf16 v16bf;
typedef __attribute__((ext_vector_type(8)))  float  v8f;

#define DEV __device__ __forceinline__

union F16x16 { uint4 u[2]; v16bf v; };
union U4x8   { uint4 u;    unsigned short s[8]; };
union U2x4   { uint2 u;    unsigned short s[4]; };

DEV v8f wmma_bf16(v16bf a, v16bf b, v8f c) {
  return __builtin_amdgcn_wmma_f32_16x16x32_bf16(
      false, a, false, b, (short)0, c, false, false);
}
DEV v8f wmma_bf16_ra(v16bf a, v16bf b, v8f c) {   // A reused by next WMMA
  return __builtin_amdgcn_wmma_f32_16x16x32_bf16(
      false, a, false, b, (short)0, c, true, false);
}

// ---- 16-lane group butterfly reductions via DPP16 (no LDS traffic) ---------
#if __has_builtin(__builtin_amdgcn_update_dpp)
DEV float grp_max16(float x) {
  int t;
  t = __builtin_amdgcn_update_dpp(0, __float_as_int(x), 0xB1, 0xf, 0xf, true);  // quad_perm xor1
  x = fmaxf(x, __int_as_float(t));
  t = __builtin_amdgcn_update_dpp(0, __float_as_int(x), 0x4E, 0xf, 0xf, true);  // quad_perm xor2
  x = fmaxf(x, __int_as_float(t));
  t = __builtin_amdgcn_update_dpp(0, __float_as_int(x), 0x141, 0xf, 0xf, true); // row_half_mirror
  x = fmaxf(x, __int_as_float(t));
  t = __builtin_amdgcn_update_dpp(0, __float_as_int(x), 0x128, 0xf, 0xf, true); // row_ror:8
  x = fmaxf(x, __int_as_float(t));
  return x;
}
DEV float grp_sum16(float x) {
  int t;
  t = __builtin_amdgcn_update_dpp(0, __float_as_int(x), 0xB1, 0xf, 0xf, true);
  x += __int_as_float(t);
  t = __builtin_amdgcn_update_dpp(0, __float_as_int(x), 0x4E, 0xf, 0xf, true);
  x += __int_as_float(t);
  t = __builtin_amdgcn_update_dpp(0, __float_as_int(x), 0x141, 0xf, 0xf, true);
  x += __int_as_float(t);
  t = __builtin_amdgcn_update_dpp(0, __float_as_int(x), 0x128, 0xf, 0xf, true);
  x += __int_as_float(t);
  return x;
}
#else
DEV float grp_max16(float x) {
  for (int mk = 1; mk < 16; mk <<= 1) x = fmaxf(x, __shfl_xor(x, mk, 32));
  return x;
}
DEV float grp_sum16(float x) {
  for (int mk = 1; mk < 16; mk <<= 1) x += __shfl_xor(x, mk, 32);
  return x;
}
#endif

// ---- async global->LDS staging (gfx1250 ASYNCcnt path) ---------------------
#define USE_ASYNC_STAGE 1
#if USE_ASYNC_STAGE
DEV void async_g2l_b128(void* lds, const void* gaddr) {
  unsigned l = (unsigned)(size_t)lds;   // generic LDS addr low 32 bits = LDS offset
  asm volatile("global_load_async_to_lds_b128 %0, %1, off"
               :: "v"(l), "v"(gaddr) : "memory");
}
DEV void wait_async0() { asm volatile("s_wait_asynccnt 0x0" ::: "memory"); }
#endif

DEV void msg_sd(const int* __restrict__ ei, int m, int& s, int& d) {
  if (m < kE) { s = ei[m]; d = ei[kE + m]; }
  else        { s = m - kE; d = m - kE; }
}

// =====================================================================
// bf16 WMMA GEMM: C[M,N] = act(A[MxK] @ B[KxN] + bias)
// Block = 128 thr (4 waves); tile 64x64; K-step 32.
// =====================================================================
template<bool HAS_BIAS, int ACT, bool OUT_F, bool OUT_B>
__global__ __launch_bounds__(128)
void gemm_bf16_kernel(const bhalf* __restrict__ A, const bhalf* __restrict__ B,
                      const float* __restrict__ bias,
                      float* __restrict__ outF, bhalf* __restrict__ outB,
                      int M, int N, int K)
{
  __shared__ bhalf sA [64 * 32];   // [row][k]
  __shared__ bhalf sBt[64 * 32];   // [col][k] transposed
  const int tid  = threadIdx.x;
  const int wave = tid >> 5;
  const int lane = tid & 31;
  const int ln   = lane & 15;
  const int lg   = lane >> 4;
  const int m0   = blockIdx.x * 64;
  const int n0   = blockIdx.y * 64;

  v8f acc[4] = {};

  for (int k0 = 0; k0 < K; k0 += 32) {
    // stage A (64x32): 2 thr/row, 16B each -> async DMA to LDS
    {
      int row = tid >> 1, half = tid & 1;
      const bhalf* src = A + (size_t)(m0 + row) * K + k0 + half * 16;
#if USE_ASYNC_STAGE
      async_g2l_b128(&sA[row * 32 + half * 16], src);
#else
      *(uint4*)&sA[row * 32 + half * 16] = *(const uint4*)src;
#endif
    }
    // stage B transposed: pair two k-rows -> b32 stores into [col][k]
    {
      int rp = tid >> 3, cg = tid & 7;
      const bhalf* p0 = B + (size_t)(k0 + 2 * rp) * N + n0 + cg * 8;
      U4x8 r0, r1;
      r0.u = *(const uint4*)p0;
      r1.u = *(const uint4*)(p0 + N);
      #pragma unroll
      for (int c = 0; c < 8; c++) {
        *(unsigned int*)&sBt[(cg * 8 + c) * 32 + 2 * rp] =
            (unsigned int)r0.s[c] | ((unsigned int)r1.s[c] << 16);
      }
    }
#if USE_ASYNC_STAGE
    wait_async0();
#endif
    __syncthreads();

    F16x16 a;
    {
      const bhalf* ap = &sA[(wave * 16 + ln) * 32 + lg * 16];
      a.u[0] = *(const uint4*)ap;
      a.u[1] = *(const uint4*)(ap + 8);
    }
    v16bf b0 = *(const v16bf*)&sBt[(0 * 16 + ln) * 32 + lg * 16];
    v16bf b1 = *(const v16bf*)&sBt[(1 * 16 + ln) * 32 + lg * 16];
    v16bf b2 = *(const v16bf*)&sBt[(2 * 16 + ln) * 32 + lg * 16];
    v16bf b3 = *(const v16bf*)&sBt[(3 * 16 + ln) * 32 + lg * 16];
    acc[0] = wmma_bf16_ra(a.v, b0, acc[0]);
    acc[1] = wmma_bf16_ra(a.v, b1, acc[1]);
    acc[2] = wmma_bf16_ra(a.v, b2, acc[2]);
    acc[3] = wmma_bf16   (a.v, b3, acc[3]);
    __syncthreads();
  }

  const int row0 = m0 + wave * 16 + lg * 8;
  #pragma unroll
  for (int ct = 0; ct < 4; ct++) {
    const int col = n0 + ct * 16 + ln;
    float bv = HAS_BIAS ? bias[col] : 0.f;
    #pragma unroll
    for (int r = 0; r < 8; r++) {
      float v = acc[ct][r] + bv;
      if (ACT == 1) v = v > 0.f ? v : 0.01f * v;
      if (ACT == 2) v = v > 0.f ? v : 0.f;
      const size_t o = (size_t)(row0 + r) * N + col;
      if (OUT_F) outF[o] = v;
      if (OUT_B) outB[o] = (bhalf)v;
    }
  }
}

// =====================================================================
// Fused flash-attention per head. Q rows stride ldq, K/V rows stride ldkv
// (supports packed QKV / KV buffers). O is bf16 [N,128].
// =====================================================================
__global__ __launch_bounds__(128)
void mha_flash_kernel(const bhalf* __restrict__ Qb, int ldq,
                      const bhalf* __restrict__ Kb, const bhalf* __restrict__ Vb,
                      int ldkv, bhalf* __restrict__ Ob, int Nn, float scale)
{
  __shared__ bhalf sK [32 * 32];    // [key][dh]
  __shared__ bhalf sVt[32 * 32];    // [dh][key]
  __shared__ bhalf sP [4][16 * 32]; // per-wave P transpose scratch
  const int tid  = threadIdx.x;
  const int wave = tid >> 5;
  const int lane = tid & 31;
  const int ln   = lane & 15;
  const int lg   = lane >> 4;
  const int hof  = blockIdx.y * kDH;
  const int qb0  = blockIdx.x * 128 + wave * 32;

  F16x16 aq[2];
  #pragma unroll
  for (int qi = 0; qi < 2; qi++) {
    const bhalf* qp = Qb + (size_t)(qb0 + qi * 16 + ln) * ldq + hof + lg * 16;
    aq[qi].u[0] = *(const uint4*)qp;
    aq[qi].u[1] = *(const uint4*)(qp + 8);
  }

  float mrow[2][8], lrow[2][8];
  v8f o0[2], o1[2];
  #pragma unroll
  for (int qi = 0; qi < 2; qi++) {
    o0[qi] = (v8f){}; o1[qi] = (v8f){};
    #pragma unroll
    for (int r = 0; r < 8; r++) { mrow[qi][r] = -1e30f; lrow[qi][r] = 0.f; }
  }

  for (int j0 = 0; j0 < Nn; j0 += 32) {
    if (j0 + 32 < Nn) {   // global_prefetch_b8 of next K/V block
      __builtin_prefetch(Kb + (size_t)(j0 + 32 + lane) * ldkv + hof, 0, 1);
      __builtin_prefetch(Vb + (size_t)(j0 + 32 + lane) * ldkv + hof, 0, 1);
    }
    // stage K row-major: async DMA to LDS
    {
      int row = tid >> 2, q = tid & 3;
      const bhalf* src = Kb + (size_t)(j0 + row) * ldkv + hof + q * 8;
#if USE_ASYNC_STAGE
      async_g2l_b128(&sK[row * 32 + q * 8], src);
#else
      *(uint4*)&sK[row * 32 + q * 8] = *(const uint4*)src;
#endif
    }
    // stage V transposed
    {
      int kp = tid >> 3, cg = tid & 7;
      const bhalf* p0 = Vb + (size_t)(j0 + 2 * kp) * ldkv + hof + cg * 4;
      U2x4 r0, r1;
      r0.u = *(const uint2*)p0;
      r1.u = *(const uint2*)(p0 + ldkv);
      #pragma unroll
      for (int c = 0; c < 4; c++) {
        *(unsigned int*)&sVt[(cg * 4 + c) * 32 + 2 * kp] =
            (unsigned int)r0.s[c] | ((unsigned int)r1.s[c] << 16);
      }
    }
#if USE_ASYNC_STAGE
    wait_async0();
#endif
    __syncthreads();

    v16bf kb0 = *(const v16bf*)&sK [ln * 32 + lg * 16];
    v16bf kb1 = *(const v16bf*)&sK [(16 + ln) * 32 + lg * 16];
    v16bf vb0 = *(const v16bf*)&sVt[ln * 32 + lg * 16];
    v16bf vb1 = *(const v16bf*)&sVt[(16 + ln) * 32 + lg * 16];

    #pragma unroll
    for (int qi = 0; qi < 2; qi++) {
      v8f zc = {};
      v8f s0 = wmma_bf16_ra(aq[qi].v, kb0, zc);
      v8f s1 = wmma_bf16   (aq[qi].v, kb1, zc);

      float p0[8], p1[8];
      #pragma unroll
      for (int r = 0; r < 8; r++) {
        float x0 = s0[r] * scale, x1 = s1[r] * scale;
        float t  = grp_max16(fmaxf(x0, x1));
        float nm  = fmaxf(mrow[qi][r], t);
        float fac = __expf(mrow[qi][r] - nm);
        mrow[qi][r] = nm;
        p0[r] = __expf(x0 - nm);
        p1[r] = __expf(x1 - nm);
        float rs = grp_sum16(p0[r] + p1[r]);
        lrow[qi][r] = lrow[qi][r] * fac + rs;
        o0[qi][r] *= fac; o1[qi][r] *= fac;
      }

      bhalf* pp = &sP[wave][0];
      #pragma unroll
      for (int r = 0; r < 8; r++) {
        pp[(lg * 8 + r) * 32 + ln]      = (bhalf)p0[r];
        pp[(lg * 8 + r) * 32 + 16 + ln] = (bhalf)p1[r];
      }
      F16x16 pa;
      {
        const bhalf* ap = pp + ln * 32 + lg * 16;
        pa.u[0] = *(const uint4*)ap;
        pa.u[1] = *(const uint4*)(ap + 8);
      }
      o0[qi] = wmma_bf16_ra(pa.v, vb0, o0[qi]);
      o1[qi] = wmma_bf16   (pa.v, vb1, o1[qi]);
    }
    __syncthreads();
  }

  #pragma unroll
  for (int qi = 0; qi < 2; qi++) {
    #pragma unroll
    for (int r = 0; r < 8; r++) {
      float inv = 1.f / (lrow[qi][r] + 1e-20f);
      const size_t row = (size_t)(qb0 + qi * 16 + lg * 8 + r) * kD + hof;
      Ob[row + ln]      = (bhalf)(o0[qi][r] * inv);
      Ob[row + 16 + ln] = (bhalf)(o1[qi][r] * inv);
    }
  }
}

// =====================================================================
// Scalar kernels
// =====================================================================
__global__ void cvt_f32_bf16_kernel(const float* __restrict__ in,
                                    bhalf* __restrict__ out, int n) {
  int i = blockIdx.x * blockDim.x + threadIdx.x;
  if (i < n) out[i] = (bhalf)in[i];
}
// pack f32 [rows,cols] -> bf16 at column offset inside [rows,ldo]
__global__ void cvt_pack_kernel(const float* __restrict__ in, bhalf* __restrict__ out,
                                int rows, int cols, int ldo, int coloff) {
  int i = blockIdx.x * blockDim.x + threadIdx.x;
  if (i >= rows * cols) return;
  int r = i / cols, c = i % cols;
  out[(size_t)r * ldo + coloff + c] = (bhalf)in[i];
}
__global__ void copy_off_f32_kernel(const float* __restrict__ in, float* __restrict__ out,
                                    int n, int off) {
  int i = blockIdx.x * blockDim.x + threadIdx.x;
  if (i < n) out[off + i] = in[i];
}
__global__ void fill_f32_kernel(float* p, float v, int n) {
  int i = blockIdx.x * blockDim.x + threadIdx.x;
  if (i < n) p[i] = v;
}
__global__ void fill_u32_kernel(unsigned* p, unsigned v, int n) {
  int i = blockIdx.x * blockDim.x + threadIdx.x;
  if (i < n) p[i] = v;
}

__global__ void fc_kernel(const float* __restrict__ x, const float* __restrict__ W,
                          const float* __restrict__ b, bhalf* __restrict__ out) {
  int idx = blockIdx.x * blockDim.x + threadIdx.x;
  if (idx >= kN * kEMB) return;
  int n = idx / kEMB, e = idx % kEMB;
  float s = b[e];
  #pragma unroll
  for (int k = 0; k < kNF; k++) s += x[n * kNF + k] * W[k * kEMB + e];
  out[idx] = (bhalf)s;
}

__global__ void gat_scores_kernel(const float* __restrict__ h,
                                  const float* __restrict__ a_src,
                                  const float* __restrict__ a_dst,
                                  float* __restrict__ al, float* __restrict__ ar, int C) {
  int idx = blockIdx.x * blockDim.x + threadIdx.x;
  if (idx >= kN * kH) return;
  int n = idx / kH, hd = idx % kH;
  const float* hp = h + (size_t)n * kH * C + hd * C;
  const float* as = a_src + hd * C;
  const float* ad = a_dst + hd * C;
  float sl = 0.f, sr = 0.f;
  for (int c = 0; c < C; c++) { float v = hp[c]; sl += v * as[c]; sr += v * ad[c]; }
  al[idx] = sl; ar[idx] = sr;
}

DEV int enc_f32(float v) {
  int iv = __float_as_int(v);
  return iv >= 0 ? iv : (iv ^ 0x7fffffff);
}
DEV float dec_f32(int e) {
  int iv = e >= 0 ? e : (e ^ 0x7fffffff);
  return __int_as_float(iv);
}

__global__ void gat_edge_max_kernel(const int* __restrict__ ei,
                                    const float* __restrict__ al,
                                    const float* __restrict__ ar,
                                    int* __restrict__ menc) {
  int m = blockIdx.x * blockDim.x + threadIdx.x;
  if (m >= kMSG) return;
  int s, d; msg_sd(ei, m, s, d);
  #pragma unroll
  for (int hd = 0; hd < kH; hd++) {
    float lg = al[s * kH + hd] + ar[d * kH + hd];
    lg = lg > 0.f ? lg : 0.2f * lg;
    atomicMax(&menc[d * kH + hd], enc_f32(lg));
  }
}

__global__ void gat_edge_sum_kernel(const int* __restrict__ ei,
                                    const float* __restrict__ al,
                                    const float* __restrict__ ar,
                                    const int* __restrict__ menc,
                                    float* __restrict__ ebuf,
                                    float* __restrict__ ssum) {
  int m = blockIdx.x * blockDim.x + threadIdx.x;
  if (m >= kMSG) return;
  int s, d; msg_sd(ei, m, s, d);
  #pragma unroll
  for (int hd = 0; hd < kH; hd++) {
    float lg = al[s * kH + hd] + ar[d * kH + hd];
    lg = lg > 0.f ? lg : 0.2f * lg;
    float e = __expf(lg - dec_f32(menc[d * kH + hd]));
    ebuf[(size_t)m * kH + hd] = e;
    atomicAdd(&ssum[d * kH + hd], e);
  }
}

__global__ void gat_aggregate_kernel(const int* __restrict__ ei,
                                     const float* __restrict__ ebuf,
                                     const float* __restrict__ ssum,
                                     const float* __restrict__ hfeat,
                                     float* __restrict__ out, int C) {
  int idx = blockIdx.x * blockDim.x + threadIdx.x;
  if (idx >= kMSG * kH) return;
  int m = idx / kH, hd = idx % kH;
  int s, d; msg_sd(ei, m, s, d);
  float alpha = ebuf[idx] / (ssum[d * kH + hd] + 1e-16f);
  const float* hp = hfeat + (size_t)s * kH * C + hd * C;
  float* op = out + (size_t)d * kH * C + hd * C;
  for (int c = 0; c < C; c++) atomicAdd(&op[c], hp[c] * alpha);
}

__global__ void gat_finish_kernel(const float* __restrict__ acc,
                                  const float* __restrict__ bias,
                                  float* __restrict__ outF, bhalf* __restrict__ outB,
                                  int total, int ncols) {
  int idx = blockIdx.x * blockDim.x + threadIdx.x;
  if (idx >= total) return;
  float v = acc[idx] + bias[idx % ncols];
  v = v > 0.f ? v : 0.01f * v;
  outF[idx] = v;
  outB[idx] = (bhalf)v;
}

template<bool OUTB>
__global__ __launch_bounds__(256)
void add_ln_kernel(const float* __restrict__ x, const float* __restrict__ y,
                   const float* __restrict__ g, const float* __restrict__ be,
                   float* __restrict__ outF, bhalf* __restrict__ outB) {
  int wave = threadIdx.x >> 5, lane = threadIdx.x & 31;
  int row = blockIdx.x * 8 + wave;
  if (row >= kN) return;
  const float* xr = x + (size_t)row * kD;
  const float* yr = y + (size_t)row * kD;
  float v[4]; float s = 0.f;
  #pragma unroll
  for (int i = 0; i < 4; i++) { int c = lane + 32 * i; v[i] = xr[c] + yr[c]; s += v[i]; }
  #pragma unroll
  for (int mk = 1; mk < 32; mk <<= 1) s += __shfl_xor(s, mk, 32);
  float mu = s * (1.f / kD);
  float q = 0.f;
  #pragma unroll
  for (int i = 0; i < 4; i++) { float d = v[i] - mu; q += d * d; }
  #pragma unroll
  for (int mk = 1; mk < 32; mk <<= 1) q += __shfl_xor(q, mk, 32);
  float rstd = rsqrtf(q * (1.f / kD) + 1e-5f);
  #pragma unroll
  for (int i = 0; i < 4; i++) {
    int c = lane + 32 * i;
    float o = (v[i] - mu) * rstd * g[c] + be[c];
    outF[(size_t)row * kD + c] = o;
    if (OUTB) outB[(size_t)row * kD + c] = (bhalf)o;
  }
}

__global__ void pool_kernel(const float* __restrict__ t, const int* __restrict__ batch,
                            float* __restrict__ pooled, float* __restrict__ cnt) {
  int idx = blockIdx.x * blockDim.x + threadIdx.x;
  if (idx >= kN * kD) return;
  int n = idx / kD, d = idx % kD;
  int b = batch[n];
  atomicAdd(&pooled[b * kD + d], t[idx]);
  if (d == 0) atomicAdd(&cnt[b], 1.f);
}

__global__ void out_kernel(const float* __restrict__ pooled, const float* __restrict__ cnt,
                           const float* __restrict__ Wout, const float* __restrict__ bout,
                           float* __restrict__ out) {
  int idx = threadIdx.x;
  if (idx >= kNG * kOUT) return;
  int g = idx / kOUT, o = idx % kOUT;
  float inv = 1.f / fmaxf(cnt[g], 1.f);
  float s = bout[o];
  for (int d = 0; d < kD; d++) s += pooled[g * kD + d] * inv * Wout[d * kOUT + o];
  out[idx] = s;
}

// ---------------- host-side helpers -----------------------------------------
template<bool HB, int ACT, bool OF, bool OB>
static inline void launch_gemm(const bhalf* A, const bhalf* B, const float* bias,
                               float* outF, bhalf* outB, int M, int N, int K,
                               hipStream_t s) {
  dim3 g(M / 64, N / 64);
  gemm_bf16_kernel<HB, ACT, OF, OB><<<g, 128, 0, s>>>(A, B, bias, outF, outB, M, N, K);
}
static inline void launch_cvt(const float* in, bhalf* out, int n, hipStream_t s) {
  cvt_f32_bf16_kernel<<<(n + 255) / 256, 256, 0, s>>>(in, out, n);
}
static inline void launch_pack(const float* in, bhalf* out, int rows, int cols, int ldo,
                               int coloff, hipStream_t s) {
  cvt_pack_kernel<<<(rows * cols + 255) / 256, 256, 0, s>>>(in, out, rows, cols, ldo, coloff);
}

extern "C" void kernel_launch(void* const* d_in, const int* in_sizes, int n_in,
                              void* d_out, int out_size, void* d_ws, size_t ws_size,
                              hipStream_t stream) {
  const float* x     = (const float*)d_in[0];
  const int*   ei    = (const int*)  d_in[1];
  const int*   batch = (const int*)  d_in[2];
  const float* W_fc  = (const float*)d_in[3];
  const float* b_fc  = (const float*)d_in[4];
  const float* W1    = (const float*)d_in[5];
  const float* b1    = (const float*)d_in[6];
  const float* a1s   = (const float*)d_in[7];
  const float* a1d   = (const float*)d_in[8];
  const float* W2    = (const float*)d_in[9];
  const float* b2    = (const float*)d_in[10];
  const float* a2s   = (const float*)d_in[11];
  const float* a2d   = (const float*)d_in[12];
  const float* Wq1   = (const float*)d_in[13]; const float* bq1 = (const float*)d_in[14];
  const float* Wk1   = (const float*)d_in[15]; const float* bk1 = (const float*)d_in[16];
  const float* Wv1   = (const float*)d_in[17]; const float* bv1 = (const float*)d_in[18];
  const float* Wo1   = (const float*)d_in[19]; const float* bo1 = (const float*)d_in[20];
  const float* Wq2   = (const float*)d_in[21]; const float* bq2 = (const float*)d_in[22];
  const float* Wk2   = (const float*)d_in[23]; const float* bk2 = (const float*)d_in[24];
  const float* Wv2   = (const float*)d_in[25]; const float* bv2 = (const float*)d_in[26];
  const float* Wo2   = (const float*)d_in[27]; const float* bo2 = (const float*)d_in[28];
  const float* Wf1   = (const float*)d_in[29]; const float* bf1 = (const float*)d_in[30];
  const float* Wf2   = (const float*)d_in[31]; const float* bf2 = (const float*)d_in[32];
  const float* g1    = (const float*)d_in[33]; const float* be1 = (const float*)d_in[34];
  const float* g2    = (const float*)d_in[35]; const float* be2 = (const float*)d_in[36];
  const float* g3    = (const float*)d_in[37]; const float* be3 = (const float*)d_in[38];
  const float* Wout  = (const float*)d_in[39]; const float* bout = (const float*)d_in[40];
  float* outp = (float*)d_out;
  (void)in_sizes; (void)n_in; (void)out_size; (void)ws_size;

  char* wp = (char*)d_ws;
  auto alloc = [&](size_t bytes) -> void* {
    void* p = wp; wp += (bytes + 255) & ~(size_t)255; return p;
  };

  bhalf* W1b    = (bhalf*)alloc((size_t)kEMB * kH * kC1 * 2);
  bhalf* W2b    = (bhalf*)alloc((size_t)(kH * kC1) * kD * 2);
  bhalf* Wqkv1b = (bhalf*)alloc((size_t)kD * 384 * 2);   // [Wq1|Wk1|Wv1]
  float* bqkv1  = (float*)alloc(384 * 4);
  bhalf* Wo1b   = (bhalf*)alloc((size_t)kD * kD * 2);
  bhalf* Wq2b   = (bhalf*)alloc((size_t)kD * kD * 2);
  bhalf* Wkv2b  = (bhalf*)alloc((size_t)kD * 256 * 2);   // [Wk2|Wv2]
  float* bkv2   = (float*)alloc(256 * 4);
  bhalf* Wo2b   = (bhalf*)alloc((size_t)kD * kD * 2);
  bhalf* Wf1b   = (bhalf*)alloc((size_t)kD * kFF * 2);
  bhalf* Wf2b   = (bhalf*)alloc((size_t)kFF * kD * 2);

  bhalf* h0b   = (bhalf*)alloc((size_t)kN * kEMB * 2);
  float* g1h   = (float*)alloc((size_t)kN * kH * kC1 * 4);
  float* al1   = (float*)alloc((size_t)kN * kH * 4);
  float* ar1   = (float*)alloc((size_t)kN * kH * 4);
  int*   m1    = (int*)  alloc((size_t)kN * kH * 4);
  float* s1    = (float*)alloc((size_t)kN * kH * 4);
  float* e1    = (float*)alloc((size_t)kMSG * kH * 4);
  float* gat1f = (float*)alloc((size_t)kN * kH * kC1 * 4);
  bhalf* gat1b = (bhalf*)alloc((size_t)kN * kH * kC1 * 2);
  float* g2h   = (float*)alloc((size_t)kN * kD * 4);
  float* al2   = (float*)alloc((size_t)kN * kH * 4);
  float* ar2   = (float*)alloc((size_t)kN * kH * 4);
  int*   m2    = (int*)  alloc((size_t)kN * kH * 4);
  float* s2    = (float*)alloc((size_t)kN * kH * 4);
  float* e2    = (float*)alloc((size_t)kMSG * kH * 4);
  float* hf    = (float*)alloc((size_t)kN * kD * 4);
  bhalf* hb    = (bhalf*)alloc((size_t)kN * kD * 2);
  bhalf* qkv1b = (bhalf*)alloc((size_t)kN * 384 * 2);    // packed Q|K|V
  bhalf* at1b  = (bhalf*)alloc((size_t)kN * kD * 2);
  float* pr1f  = (float*)alloc((size_t)kN * kD * 4);
  float* t1f   = (float*)alloc((size_t)kN * kD * 4);
  bhalf* t1b   = (bhalf*)alloc((size_t)kN * kD * 2);
  bhalf* q2b   = (bhalf*)alloc((size_t)kN * kD * 2);
  bhalf* kv2b  = (bhalf*)alloc((size_t)kN * 256 * 2);    // packed K|V
  bhalf* at2b  = (bhalf*)alloc((size_t)kN * kD * 2);
  float* pr2f  = (float*)alloc((size_t)kN * kD * 4);
  float* t2f   = (float*)alloc((size_t)kN * kD * 4);
  bhalf* t2b   = (bhalf*)alloc((size_t)kN * kD * 2);
  bhalf* ff1b  = (bhalf*)alloc((size_t)kN * kFF * 2);
  float* ff2f  = (float*)alloc((size_t)kN * kD * 4);
  float* t3f   = (float*)alloc((size_t)kN * kD * 4);
  float* pooled= (float*)alloc((size_t)kNG * kD * 4);
  float* cnt   = (float*)alloc((size_t)kNG * 4);

  // ---- weight conversions / packing ----
  launch_cvt(W1, W1b, kEMB * kH * kC1, stream);
  launch_cvt(W2, W2b, kH * kC1 * kD, stream);
  launch_pack(Wq1, Wqkv1b, kD, kD, 384, 0,   stream);
  launch_pack(Wk1, Wqkv1b, kD, kD, 384, 128, stream);
  launch_pack(Wv1, Wqkv1b, kD, kD, 384, 256, stream);
  copy_off_f32_kernel<<<1, 128, 0, stream>>>(bq1, bqkv1, kD, 0);
  copy_off_f32_kernel<<<1, 128, 0, stream>>>(bk1, bqkv1, kD, 128);
  copy_off_f32_kernel<<<1, 128, 0, stream>>>(bv1, bqkv1, kD, 256);
  launch_cvt(Wo1, Wo1b, kD * kD, stream);
  launch_cvt(Wq2, Wq2b, kD * kD, stream);
  launch_pack(Wk2, Wkv2b, kD, kD, 256, 0,   stream);
  launch_pack(Wv2, Wkv2b, kD, kD, 256, 128, stream);
  copy_off_f32_kernel<<<1, 128, 0, stream>>>(bk2, bkv2, kD, 0);
  copy_off_f32_kernel<<<1, 128, 0, stream>>>(bv2, bkv2, kD, 128);
  launch_cvt(Wo2, Wo2b, kD * kD, stream);
  launch_cvt(Wf1, Wf1b, kD * kFF, stream);
  launch_cvt(Wf2, Wf2b, kFF * kD, stream);

  // ---- input FC ----
  fc_kernel<<<(kN * kEMB + 255) / 256, 256, 0, stream>>>(x, W_fc, b_fc, h0b);

  // ---- GAT layer 1 (C=64) ----
  launch_gemm<false, 0, true, false>(h0b, W1b, nullptr, g1h, nullptr, kN, kH * kC1, kEMB, stream);
  gat_scores_kernel<<<(kN * kH + 255) / 256, 256, 0, stream>>>(g1h, a1s, a1d, al1, ar1, kC1);
  fill_u32_kernel<<<(kN * kH + 255) / 256, 256, 0, stream>>>((unsigned*)m1, 0x80000000u, kN * kH);
  fill_f32_kernel<<<(kN * kH + 255) / 256, 256, 0, stream>>>(s1, 0.f, kN * kH);
  fill_f32_kernel<<<(kN * kH * kC1 + 255) / 256, 256, 0, stream>>>(gat1f, 0.f, kN * kH * kC1);
  gat_edge_max_kernel<<<(kMSG + 255) / 256, 256, 0, stream>>>(ei, al1, ar1, m1);
  gat_edge_sum_kernel<<<(kMSG + 255) / 256, 256, 0, stream>>>(ei, al1, ar1, m1, e1, s1);
  gat_aggregate_kernel<<<(kMSG * kH + 255) / 256, 256, 0, stream>>>(ei, e1, s1, g1h, gat1f, kC1);
  gat_finish_kernel<<<(kN * kH * kC1 + 255) / 256, 256, 0, stream>>>(
      gat1f, b1, gat1f, gat1b, kN * kH * kC1, kH * kC1);

  // ---- GAT layer 2 (C=32) ----
  launch_gemm<false, 0, true, false>(gat1b, W2b, nullptr, g2h, nullptr, kN, kD, kH * kC1, stream);
  gat_scores_kernel<<<(kN * kH + 255) / 256, 256, 0, stream>>>(g2h, a2s, a2d, al2, ar2, kC2);
  fill_u32_kernel<<<(kN * kH + 255) / 256, 256, 0, stream>>>((unsigned*)m2, 0x80000000u, kN * kH);
  fill_f32_kernel<<<(kN * kH + 255) / 256, 256, 0, stream>>>(s2, 0.f, kN * kH);
  fill_f32_kernel<<<(kN * kD + 255) / 256, 256, 0, stream>>>(hf, 0.f, kN * kD);
  gat_edge_max_kernel<<<(kMSG + 255) / 256, 256, 0, stream>>>(ei, al2, ar2, m2);
  gat_edge_sum_kernel<<<(kMSG + 255) / 256, 256, 0, stream>>>(ei, al2, ar2, m2, e2, s2);
  gat_aggregate_kernel<<<(kMSG * kH + 255) / 256, 256, 0, stream>>>(ei, e2, s2, g2h, hf, kC2);
  gat_finish_kernel<<<(kN * kD + 255) / 256, 256, 0, stream>>>(hf, b2, hf, hb, kN * kD, kD);

  const float scale = 0.17677669529663689f;  // 1/sqrt(32)
  dim3 fgrid(kN / 128, kH);

  // ---- transformer: self-attention (fused QKV projection) ----
  launch_gemm<true, 0, false, true>(hb, Wqkv1b, bqkv1, nullptr, qkv1b, kN, 384, kD, stream);
  mha_flash_kernel<<<fgrid, 128, 0, stream>>>(qkv1b + 0, 384, qkv1b + 128, qkv1b + 256, 384,
                                              at1b, kN, scale);
  launch_gemm<true, 0, true, false>(at1b, Wo1b, bo1, pr1f, nullptr, kN, kD, kD, stream);
  add_ln_kernel<true><<<kN / 8, 256, 0, stream>>>(hf, pr1f, g1, be1, t1f, t1b);

  // ---- transformer: cross-attention (memory = GAT output h) ----
  launch_gemm<true, 0, false, true>(t1b, Wq2b, bq2, nullptr, q2b, kN, kD, kD, stream);
  launch_gemm<true, 0, false, true>(hb, Wkv2b, bkv2, nullptr, kv2b, kN, 256, kD, stream);
  mha_flash_kernel<<<fgrid, 128, 0, stream>>>(q2b, kD, kv2b + 0, kv2b + 128, 256,
                                              at2b, kN, scale);
  launch_gemm<true, 0, true, false>(at2b, Wo2b, bo2, pr2f, nullptr, kN, kD, kD, stream);
  add_ln_kernel<true><<<kN / 8, 256, 0, stream>>>(t1f, pr2f, g2, be2, t2f, t2b);

  // ---- feed-forward ----
  launch_gemm<true, 2, false, true>(t2b, Wf1b, bf1, nullptr, ff1b, kN, kFF, kD, stream);
  launch_gemm<true, 0, true, false>(ff1b, Wf2b, bf2, ff2f, nullptr, kN, kD, kFF, stream);
  add_ln_kernel<false><<<kN / 8, 256, 0, stream>>>(t2f, ff2f, g3, be3, t3f, nullptr);

  // ---- pool + output head ----
  fill_f32_kernel<<<(kNG * kD + 255) / 256, 256, 0, stream>>>(pooled, 0.f, kNG * kD);
  fill_f32_kernel<<<1, 64, 0, stream>>>(cnt, 0.f, kNG);
  pool_kernel<<<(kN * kD + 255) / 256, 256, 0, stream>>>(t3f, batch, pooled, cnt);
  out_kernel<<<1, 64, 0, stream>>>(pooled, cnt, Wout, bout, outp);
}
